// CausalSelfAttention_44049184588425
// MI455X (gfx1250) — matrix operand order repaired and checked by reading, not compiled, not run on hardware
//
#include <hip/hip_runtime.h>

// ---------------------------------------------------------------------------
// CDNA5 (gfx1250) causal self-attention: bf16 WMMA everywhere, f32 accum.
// Round 2: software-pipelined GEMM, clustered operand loads in attention,
// packed b32 V-transpose staging.
// ---------------------------------------------------------------------------

typedef __attribute__((ext_vector_type(16))) __bf16 v16bf;
typedef __attribute__((ext_vector_type(8)))  float  v8f;
typedef __attribute__((ext_vector_type(4)))  unsigned int u32x4;

union BF16x16 { v16bf v; u32x4 u[2]; };          // one WMMA 16-bit operand per lane
union V32     { u32x4 u[4]; unsigned short h[32]; };

#define TSEQ 4096
#define CQKV 2304
#define DMODEL 768

__device__ __forceinline__ unsigned short f2bfu(float f) {
  unsigned u = __float_as_uint(f);
  u += 0x7fffu + ((u >> 16) & 1u);               // round-to-nearest-even
  return (unsigned short)(u >> 16);
}

__device__ __forceinline__ v8f wmma_bf16(v16bf a, v16bf b, v8f c) {
  // D = A(16x32 bf16) * B(32x16 bf16) + C(16x16 f32)
  return __builtin_amdgcn_wmma_f32_16x16x32_bf16(false, a, false, b, (short)0, c,
                                                 false, false);
}

// ---------------------------------------------------------------------------
// fp32 -> bf16 elementwise
// ---------------------------------------------------------------------------
__global__ void k_f32_to_bf16(const float* __restrict__ in,
                              unsigned short* __restrict__ out, int n) {
  int i = blockIdx.x * 256 + threadIdx.x;
  if (i < n) out[i] = f2bfu(in[i]);
}

// ---------------------------------------------------------------------------
// W[K][N] fp32 -> Wt[N][K] bf16  (so WMMA B-operand reads are contiguous)
// ---------------------------------------------------------------------------
__global__ void k_transpose_to_bf16(const float* __restrict__ W,
                                    unsigned short* __restrict__ Wt,
                                    int K, int N) {
  int i = blockIdx.x * 256 + threadIdx.x;
  if (i >= K * N) return;
  int k = i / N, n = i % N;
  Wt[(size_t)n * K + k] = f2bfu(W[i]);
}

// ---------------------------------------------------------------------------
// Wave-tiled GEMM: C[M,N] = A[M,K](bf16) * Bt[N,K](bf16)^T + bias
// Each wave32 computes a 16x64 tile with 4 f32 accumulators.
// Double-buffered: next k-step's operands are loaded before current WMMAs.
// ---------------------------------------------------------------------------
template<bool OUT_BF16>
__global__ __launch_bounds__(256) void k_gemm_bf16(
    const unsigned short* __restrict__ A, const unsigned short* __restrict__ Bt,
    const float* __restrict__ bias, unsigned short* __restrict__ Cb,
    float* __restrict__ Cf, int M, int N, int K)
{
  int lane = threadIdx.x & 31;
  int wid  = (blockIdx.x * blockDim.x + threadIdx.x) >> 5;
  int ntN  = N >> 6;
  int mt = wid / ntN, nt = wid % ntN;
  int m0 = mt << 4, n0 = nt << 6;
  if (m0 >= M) return;
  int r = lane & 15, g = lane >> 4;

  v8f zero = {0.f, 0.f, 0.f, 0.f, 0.f, 0.f, 0.f, 0.f};
  v8f acc[4];
  #pragma unroll
  for (int i = 0; i < 4; i++) acc[i] = zero;

  // A: row = r, K chunk at 16g. B: col = r, K chunk at 16g.
  const unsigned short* arow = A + (size_t)(m0 + r) * K + 16 * g;
  const unsigned short* brow[4];
  #pragma unroll
  for (int nb = 0; nb < 4; nb++)
    brow[nb] = Bt + (size_t)(n0 + nb * 16 + r) * K + 16 * g;

  BF16x16 a_cur, b_cur[4];
  a_cur.u[0] = *(const u32x4*)(arow);
  a_cur.u[1] = *(const u32x4*)(arow + 8);
  #pragma unroll
  for (int nb = 0; nb < 4; nb++) {
    b_cur[nb].u[0] = *(const u32x4*)(brow[nb]);
    b_cur[nb].u[1] = *(const u32x4*)(brow[nb] + 8);
  }

  int k0 = 0;
  for (; k0 + 32 < K; k0 += 32) {
    // issue next tile's loads first -> they fly while WMMAs execute
    BF16x16 a_nxt, b_nxt[4];
    a_nxt.u[0] = *(const u32x4*)(arow + k0 + 32);
    a_nxt.u[1] = *(const u32x4*)(arow + k0 + 40);
    #pragma unroll
    for (int nb = 0; nb < 4; nb++) {
      b_nxt[nb].u[0] = *(const u32x4*)(brow[nb] + k0 + 32);
      b_nxt[nb].u[1] = *(const u32x4*)(brow[nb] + k0 + 40);
    }
    #pragma unroll
    for (int nb = 0; nb < 4; nb++)
      acc[nb] = wmma_bf16(a_cur.v, b_cur[nb].v, acc[nb]);
    a_cur = a_nxt;
    #pragma unroll
    for (int nb = 0; nb < 4; nb++) b_cur[nb] = b_nxt[nb];
  }
  #pragma unroll
  for (int nb = 0; nb < 4; nb++)
    acc[nb] = wmma_bf16(a_cur.v, b_cur[nb].v, acc[nb]);

  #pragma unroll
  for (int nb = 0; nb < 4; nb++) {
    int n = n0 + nb * 16 + r;
    float bv = bias[n];
    #pragma unroll
    for (int v = 0; v < 8; v++) {
      int row = m0 + v + 8 * g;                  // C layout: M = vgpr + 8*half
      float val = acc[nb][v] + bv;
      if (OUT_BF16) Cb[(size_t)row * N + n] = f2bfu(val);
      else          Cf[(size_t)row * N + n] = val;
    }
  }
}

// ---------------------------------------------------------------------------
// Flash attention: one wave per 16-query tile per (batch,head).
// qkv is bf16 [B*T][2304]: q at col 0, k at 768, v at 1536 (+h*64 per head).
// ---------------------------------------------------------------------------
__global__ __launch_bounds__(128) void k_attention(
    const unsigned short* __restrict__ qkv, unsigned short* __restrict__ yatt)
{
  __shared__ unsigned short lds_vt[4][64 * 32];  // per-wave V^T tile: [d][key]
  __shared__ unsigned short lds_p [4][16 * 32];  // per-wave P relayout buffer

  int lane = threadIdx.x & 31;
  int w    = threadIdx.x >> 5;
  int gwid = blockIdx.x * 4 + w;                 // 0 .. 24*256-1
  int bh = gwid >> 8;                            // 0..23
  int qt = gwid & 255;
  int b = bh / 12, h = bh % 12;
  int q0 = qt << 4;
  int r = lane & 15, g = lane >> 4;

  const unsigned short* base = qkv + (size_t)b * TSEQ * CQKV + h * 64;
  const unsigned short* Qp = base;
  const unsigned short* Kp = base + 768;
  const unsigned short* Vp = base + 1536;

  // Q in A-operand layout, held in registers for the whole tile.
  const unsigned short* qrow = Qp + (size_t)(q0 + r) * CQKV;
  BF16x16 aQ0, aQ1;
  aQ0.u[0] = *(const u32x4*)(qrow + 16 * g);       // d  0..31
  aQ0.u[1] = *(const u32x4*)(qrow + 8 + 16 * g);
  aQ1.u[0] = *(const u32x4*)(qrow + 32 + 16 * g);  // d 32..63
  aQ1.u[1] = *(const u32x4*)(qrow + 40 + 16 * g);

  float mrow[8], srow[8];
  #pragma unroll
  for (int v = 0; v < 8; v++) { mrow[v] = -1e30f; srow[v] = 0.f; }
  v8f zero = {0.f, 0.f, 0.f, 0.f, 0.f, 0.f, 0.f, 0.f};
  v8f yacc[4];
  #pragma unroll
  for (int nb = 0; nb < 4; nb++) yacc[nb] = zero;

  // V staging mapping: lane owns 2 adjacent key rows (kk, kk+1) and a 32-wide
  // d-range, so transposed writes pack two keys per b32 store.
  int kk = 2 * r;            // key pair within tile
  int d0 = 32 * g;           // d half

  const float scale = 0.125f;                    // 1/sqrt(64)
  int kend = q0 + 16;                            // exclusive causal bound
  for (int kt = 0; kt < kend; kt += 32) {
    // ---- issue all global loads for this tile first (K operands + V rows)
    const unsigned short* krow0 = Kp + (size_t)(kt + r) * CQKV + 16 * g;
    const unsigned short* krow1 = Kp + (size_t)(kt + 16 + r) * CQKV + 16 * g;
    BF16x16 bK[4];
    bK[0].u[0] = *(const u32x4*)(krow0);          // keys kt..kt+15, d 0..31
    bK[0].u[1] = *(const u32x4*)(krow0 + 8);
    bK[1].u[0] = *(const u32x4*)(krow0 + 32);     // keys kt..kt+15, d 32..63
    bK[1].u[1] = *(const u32x4*)(krow0 + 40);
    bK[2].u[0] = *(const u32x4*)(krow1);          // keys kt+16..kt+31, d 0..31
    bK[2].u[1] = *(const u32x4*)(krow1 + 8);
    bK[3].u[0] = *(const u32x4*)(krow1 + 32);     // keys kt+16..kt+31, d 32..63
    bK[3].u[1] = *(const u32x4*)(krow1 + 40);

    const unsigned short* vrow0 = Vp + (size_t)(kt + kk) * CQKV + d0;
    const unsigned short* vrow1 = Vp + (size_t)(kt + kk + 1) * CQKV + d0;
    V32 v0, v1;
    #pragma unroll
    for (int j = 0; j < 4; j++) v0.u[j] = *(const u32x4*)(vrow0 + 8 * j);
    #pragma unroll
    for (int j = 0; j < 4; j++) v1.u[j] = *(const u32x4*)(vrow1 + 8 * j);

    if (kt + 32 < kend)
      __builtin_prefetch(Vp + (size_t)(kt + 32 + lane) * CQKV, 0, 0);

    // ---- stage V^T tile into this wave's LDS region (packed 2 keys / b32)
    #pragma unroll
    for (int j = 0; j < 32; j++) {
      unsigned int pk = (unsigned int)v0.h[j] | ((unsigned int)v1.h[j] << 16);
      *(unsigned int*)&lds_vt[w][(d0 + j) * 32 + kk] = pk;
    }

    // ---- S = Q K^T: two independent accumulator chains, loads already done
    v8f sacc[2];
    sacc[0] = wmma_bf16(aQ0.v, bK[0].v, zero);
    sacc[1] = wmma_bf16(aQ0.v, bK[2].v, zero);
    sacc[0] = wmma_bf16(aQ1.v, bK[1].v, sacc[0]);
    sacc[1] = wmma_bf16(aQ1.v, bK[3].v, sacc[1]);

    // ---- causal mask + online softmax update; emit P (bf16) into LDS
    #pragma unroll
    for (int v = 0; v < 8; v++) {
      int row = q0 + v + 8 * g;
      float s0 = sacc[0][v] * scale;
      float s1 = sacc[1][v] * scale;
      if (kt + r > row)      s0 = -1e30f;
      if (kt + 16 + r > row) s1 = -1e30f;
      float cm = fmaxf(s0, s1);
      #pragma unroll
      for (int off = 1; off < 16; off <<= 1)
        cm = fmaxf(cm, __shfl_xor(cm, off, 32));  // stays within 16-lane half
      float mn    = fmaxf(mrow[v], cm);
      float alpha = __expf(mrow[v] - mn);
      float p0 = __expf(s0 - mn);
      float p1 = __expf(s1 - mn);
      float rs = p0 + p1;
      #pragma unroll
      for (int off = 1; off < 16; off <<= 1) rs += __shfl_xor(rs, off, 32);
      srow[v] = srow[v] * alpha + rs;
      mrow[v] = mn;
      #pragma unroll
      for (int nb = 0; nb < 4; nb++) yacc[nb][v] *= alpha;
      int rr = v + 8 * g;
      lds_p[w][rr * 32 + r]      = f2bfu(p0);
      lds_p[w][rr * 32 + 16 + r] = f2bfu(p1);
    }

    // ---- y += P * V  (same-wave LDS is in-order; no block barrier needed)
    BF16x16 aP;
    const unsigned short* prow = &lds_p[w][r * 32];
    aP.u[0] = *(const u32x4*)(prow + 16 * g);
    aP.u[1] = *(const u32x4*)(prow + 8 + 16 * g);
    BF16x16 bV[4];
    #pragma unroll
    for (int nb = 0; nb < 4; nb++) {
      const unsigned short* vt = &lds_vt[w][(nb * 16 + r) * 32 + 16 * g];
      bV[nb].u[0] = *(const u32x4*)(vt);
      bV[nb].u[1] = *(const u32x4*)(vt + 8);
    }
    #pragma unroll
    for (int nb = 0; nb < 4; nb++)
      yacc[nb] = wmma_bf16(aP.v, bV[nb].v, yacc[nb]);
  }

  // ---- normalize and store y_att (bf16, [B*T][768], heads concatenated)
  #pragma unroll
  for (int v = 0; v < 8; v++) {
    float inv = 1.f / srow[v];
    size_t rowoff = (size_t)(b * TSEQ + q0 + v + 8 * g) * DMODEL + h * 64;
    #pragma unroll
    for (int nb = 0; nb < 4; nb++)
      yatt[rowoff + nb * 16 + r] = f2bfu(yacc[nb][v] * inv);
  }
}

// ---------------------------------------------------------------------------
// Launcher
// ---------------------------------------------------------------------------
extern "C" void kernel_launch(void* const* d_in, const int* in_sizes, int n_in,
                              void* d_out, int out_size, void* d_ws, size_t ws_size,
                              hipStream_t stream) {
  (void)in_sizes; (void)n_in; (void)out_size; (void)ws_size;
  const float* x     = (const float*)d_in[0];
  const float* Wqkv  = (const float*)d_in[1];
  const float* bqkv  = (const float*)d_in[2];
  const float* Wproj = (const float*)d_in[3];
  const float* bproj = (const float*)d_in[4];
  float* out = (float*)d_out;

  const int M  = 2 * TSEQ;        // 8192 token rows
  const int C  = DMODEL;          // 768
  const int N1 = CQKV;            // 2304

  char* ws = (char*)d_ws;
  auto align256 = [](size_t s) { return (s + 255) & ~(size_t)255; };
  size_t off = 0;
  unsigned short* xb     = (unsigned short*)(ws + off); off += align256((size_t)M * C * 2);
  unsigned short* wqkvT  = (unsigned short*)(ws + off); off += align256((size_t)N1 * C * 2);
  unsigned short* wprojT = (unsigned short*)(ws + off); off += align256((size_t)C * C * 2);
  unsigned short* qkvb   = (unsigned short*)(ws + off); off += align256((size_t)M * N1 * 2);
  unsigned short* yatt   = (unsigned short*)(ws + off); off += align256((size_t)M * C * 2);

  int nx = M * C;
  k_f32_to_bf16<<<(nx + 255) / 256, 256, 0, stream>>>(x, xb, nx);
  int nw1 = C * N1;
  k_transpose_to_bf16<<<(nw1 + 255) / 256, 256, 0, stream>>>(Wqkv, wqkvT, C, N1);
  int nw2 = C * C;
  k_transpose_to_bf16<<<(nw2 + 255) / 256, 256, 0, stream>>>(Wproj, wprojT, C, C);

  // GEMM1: qkv(bf16) = x @ W_qkv + b   -> 512*36 = 18432 wave tiles
  int waves1 = (M / 16) * (N1 / 64);
  k_gemm_bf16<true><<<waves1 / 8, 256, 0, stream>>>(xb, wqkvT, bqkv, qkvb,
                                                    nullptr, M, N1, C);

  // Flash attention: 24 heads * 256 q-tiles = 6144 waves, 4 waves/block
  k_attention<<<(24 * 256) / 4, 128, 0, stream>>>(qkvb, yatt);

  // GEMM2: out(f32) = y_att @ W_proj + b -> 512*12 = 6144 wave tiles
  int waves2 = (M / 16) * (C / 64);
  k_gemm_bf16<false><<<waves2 / 8, 256, 0, stream>>>(yatt, wprojT, bproj,
                                                     nullptr, out, M, C, C);
}
